// Attention_11269994184820
// MI455X (gfx1250) — matrix operand order repaired and checked by reading, not compiled, not run on hardware
//
#include <hip/hip_runtime.h>
#include <hip/hip_bf16.h>

// ---------------------------------------------------------------------------
// PVT spatial-reduction attention on MI455X (gfx1250, wave32, WMMA).
// B=4, N=4096 (64x64), C=256, heads=8, dh=dv=32, sr=2 -> Nk=1024.
// Memory-bound on relative_pos (134 MB f32) -> flash-attention fusion so the
// 512 MB score tensor never hits HBM; all GEMMs via v_wmma_f32_16x16x32_f16.
// ---------------------------------------------------------------------------

typedef __attribute__((ext_vector_type(16))) _Float16 v16h;
typedef __attribute__((ext_vector_type(8)))  _Float16 v8h;
typedef __attribute__((ext_vector_type(8)))  float    v8f;

#define B_SZ   4
#define N_TOK  4096
#define NK_TOK 1024
#define C_DIM  256
#define HEADS  8
#define DH     32

__device__ __forceinline__ v16h cat8(v8h a, v8h b) {
  return __builtin_shufflevector(a, b, 0,1,2,3,4,5,6,7,8,9,10,11,12,13,14,15);
}
__device__ __forceinline__ v8f wmma_f16(v16h a, v16h b, v8f c) {
  // D = A(16x32 f16) x B(32x16 f16) + C(16x16 f32)
  return __builtin_amdgcn_wmma_f32_16x16x32_f16(false, a, false, b, (short)0, c,
                                                false, false);
}

// ---------------------------------------------------------------------- prep
__global__ void cvt_f32_f16(const float* __restrict__ in,
                            _Float16* __restrict__ out, int n) {
  int i = blockIdx.x * blockDim.x + threadIdx.x;
  int stride = gridDim.x * blockDim.x;
  for (; i < n; i += stride) out[i] = (_Float16)in[i];
}

// depthwise 2x2 stride-2 conv + BatchNorm(eval) -> xk [B, Nk, C] f16
__global__ void conv_bn(const float* __restrict__ x, const float* __restrict__ srw,
                        const float* __restrict__ srb, const float* __restrict__ g,
                        const float* __restrict__ bta, const float* __restrict__ mu,
                        const float* __restrict__ var, _Float16* __restrict__ xkh) {
  int e = blockIdx.x * blockDim.x + threadIdx.x;   // B*Nk*C = 1,048,576 exact
  int c = e & (C_DIM - 1);
  int t = (e >> 8) & (NK_TOK - 1);
  int b = e >> 18;
  int oh = t >> 5, ow = t & 31;                    // 32x32 reduced grid
  const float* xb = x + (size_t)b * N_TOK * C_DIM;
  float acc = srb[c];
#pragma unroll
  for (int i = 0; i < 2; ++i)
#pragma unroll
    for (int j = 0; j < 2; ++j) {
      int n = (2 * oh + i) * 64 + (2 * ow + j);
      acc += xb[(size_t)n * C_DIM + c] * srw[c * 4 + i * 2 + j];
    }
  float inv_std = g[c] * rsqrtf(var[c] + 1e-5f);
  xkh[e] = (_Float16)((acc - mu[c]) * inv_std + bta[c]);
}

// ---------------------------------------------------------------- WMMA GEMM
// out[m,n] = scale * sum_k A[m,k] * W[n,k]  (+bias, f32 out, when F32OUT)
// grid = (M/16, C/16/4), block = 128 (4 waves, one 16x16 tile per wave)
template <bool F32OUT>
__global__ void gemm_wmma(const _Float16* __restrict__ A,
                          const _Float16* __restrict__ W,
                          _Float16* __restrict__ out16, float* __restrict__ out32,
                          const float* __restrict__ bias, float scale) {
  const int lane  = threadIdx.x & 31;
  const int wave  = threadIdx.x >> 5;
  const int mtile = blockIdx.x;
  const int ntile = blockIdx.y * 4 + wave;
  const int l15   = lane & 15;
  const int kbA   = (lane < 16) ? 0 : 16;   // contiguous k = 16h + s (matches B)
  const int kbB   = (lane < 16) ? 0 : 16;   // contiguous k = 16h + s (matches A)
  const int mrow  = mtile * 16 + l15;
  const int ncol  = ntile * 16 + l15;
  const _Float16* arow = A + (size_t)mrow * C_DIM;
  const _Float16* wrow = W + (size_t)ncol * C_DIM;
  v8f acc = {};
#pragma unroll
  for (int kc = 0; kc < C_DIM; kc += 32) {
    v16h a = cat8(*(const v8h*)(arow + kc + kbA),
                  *(const v8h*)(arow + kc + kbA + 8));
    v16h b = cat8(*(const v8h*)(wrow + kc + kbB),
                  *(const v8h*)(wrow + kc + kbB + 8));
    acc = wmma_f16(a, b, acc);
  }
  const int rbase = (lane < 16) ? 0 : 8;    // C/D layout: vgpr r = row r / 8+r
#pragma unroll
  for (int r = 0; r < 8; ++r) {
    int row = mtile * 16 + rbase + r;
    float v = acc[r] * scale;
    if (F32OUT)
      out32[(size_t)row * C_DIM + ncol] = v + bias[ncol];
    else
      out16[(size_t)row * C_DIM + ncol] = (_Float16)v;
  }
}

// ---------------------------------------------------- flash attention (WMMA)
// grid = (N/64, B*heads), block = 128 (4 waves, one 16-row N-tile per wave)
__global__ void attn_wmma(const _Float16* __restrict__ Qh,
                          const _Float16* __restrict__ Kh,
                          const _Float16* __restrict__ Vh,
                          const float* __restrict__ relpos,
                          _Float16* __restrict__ Oh) {
  __shared__ _Float16 sK[32 * 32];        // [token][dh]
  __shared__ _Float16 sVT[32 * 32];       // [dh][token] (transposed)
  __shared__ _Float16 sP[4 * 16 * 32];    // per-wave P scratch (layout shuffle)

  const int lane  = threadIdx.x & 31;
  const int wave  = threadIdx.x >> 5;
  const int bh    = blockIdx.y;
  const int b     = bh >> 3, h = bh & 7;
  const int ntile = blockIdx.x * 4 + wave;           // 0..255
  const int hoff  = h * DH;
  const int l15   = lane & 15;
  const int kbA   = (lane < 16) ? 0 : 16;
  const int kbB   = (lane < 16) ? 0 : 16;
  const int rbase = (lane < 16) ? 0 : 8;

  // Q fragment (A layout), already scaled by dh^-0.5 in the Q GEMM
  const _Float16* qrow = Qh + ((size_t)(b * N_TOK + ntile * 16 + l15) * C_DIM + hoff);
  v16h qf = cat8(*(const v8h*)(qrow + kbA), *(const v8h*)(qrow + kbA + 8));

  v8f o0 = {}, o1 = {};
  float m_run[8], l_run[8];
#pragma unroll
  for (int r = 0; r < 8; ++r) { m_run[r] = -3.0e38f; l_run[r] = 0.0f; }

  const int tok  = threadIdx.x >> 2;   // 0..31 (cooperative K/V staging)
  const int part = threadIdx.x & 3;    // 0..3
  _Float16* sPw = sP + wave * 512;

  for (int kk = 0; kk < NK_TOK; kk += 32) {
    __syncthreads();
    // stage K chunk [32 tok x 32 dh] and V^T chunk [32 dh x 32 tok]
    {
      size_t gbase = (size_t)(b * NK_TOK + kk + tok) * C_DIM + hoff + part * 8;
      v8h kv = *(const v8h*)(Kh + gbase);
      *(v8h*)(sK + tok * 32 + part * 8) = kv;
      v8h vv = *(const v8h*)(Vh + gbase);
#pragma unroll
      for (int j = 0; j < 8; ++j) sVT[(part * 8 + j) * 32 + tok] = vv[j];
    }
    __syncthreads();

    // scores: S = Q x K^T for two 16-col token tiles (K dim == dh == 32)
    v16h kb0 = cat8(*(const v8h*)(sK + l15 * 32 + kbB),
                    *(const v8h*)(sK + l15 * 32 + kbB + 8));
    v16h kb1 = cat8(*(const v8h*)(sK + (16 + l15) * 32 + kbB),
                    *(const v8h*)(sK + (16 + l15) * 32 + kbB + 8));
    v8f s0 = {}, s1 = {};
    s0 = wmma_f16(qf, kb0, s0);
    s1 = wmma_f16(qf, kb1, s1);

    // + relative position bias (the dominant HBM stream, coalesced per row)
    const float* rp = relpos +
        ((size_t)h * N_TOK + ntile * 16 + rbase) * NK_TOK + kk + l15;
#pragma unroll
    for (int r = 0; r < 8; ++r) {
      s0[r] += rp[(size_t)r * NK_TOK];
      s1[r] += rp[(size_t)r * NK_TOK + 16];
    }

    // online softmax; rows live across 16-lane groups in the C layout
#pragma unroll
    for (int r = 0; r < 8; ++r) {
      float tm = fmaxf(s0[r], s1[r]);
      tm = fmaxf(tm, __shfl_xor(tm, 1));
      tm = fmaxf(tm, __shfl_xor(tm, 2));
      tm = fmaxf(tm, __shfl_xor(tm, 4));
      tm = fmaxf(tm, __shfl_xor(tm, 8));
      float mn    = fmaxf(m_run[r], tm);
      float alpha = __expf(m_run[r] - mn);
      float p0 = __expf(s0[r] - mn);
      float p1 = __expf(s1[r] - mn);
      float rs = p0 + p1;
      rs += __shfl_xor(rs, 1);
      rs += __shfl_xor(rs, 2);
      rs += __shfl_xor(rs, 4);
      rs += __shfl_xor(rs, 8);
      l_run[r] = l_run[r] * alpha + rs;
      m_run[r] = mn;
      o0[r] *= alpha;
      o1[r] *= alpha;
      sPw[(rbase + r) * 32 + l15]      = (_Float16)p0;   // C-layout -> LDS
      sPw[(rbase + r) * 32 + 16 + l15] = (_Float16)p1;
    }
    // same-wave LDS RAW: all lanes' P stores must land before A-layout reload
    asm volatile("s_wait_dscnt 0" ::: "memory");

    // reload P as a 16x32 A fragment, accumulate O += P x V
    const _Float16* prow = sPw + l15 * 32 + kbA;
    v16h pf  = cat8(*(const v8h*)prow, *(const v8h*)(prow + 8));
    v16h vb0 = cat8(*(const v8h*)(sVT + l15 * 32 + kbB),
                    *(const v8h*)(sVT + l15 * 32 + kbB + 8));
    v16h vb1 = cat8(*(const v8h*)(sVT + (16 + l15) * 32 + kbB),
                    *(const v8h*)(sVT + (16 + l15) * 32 + kbB + 8));
    o0 = wmma_f16(pf, vb0, o0);
    o1 = wmma_f16(pf, vb1, o1);
  }

  // finalize: divide by softmax denominator, write f16 [B, N, C]
#pragma unroll
  for (int r = 0; r < 8; ++r) {
    float inv = 1.0f / l_run[r];
    int row = ntile * 16 + rbase + r;
    _Float16* op = Oh + ((size_t)(b * N_TOK + row) * C_DIM + hoff);
    op[l15]      = (_Float16)(o0[r] * inv);
    op[16 + l15] = (_Float16)(o1[r] * inv);
  }
}

// ------------------------------------------------------------------- launch
extern "C" void kernel_launch(void* const* d_in, const int* in_sizes, int n_in,
                              void* d_out, int out_size, void* d_ws, size_t ws_size,
                              hipStream_t stream) {
  const float* x      = (const float*)d_in[0];
  const float* relpos = (const float*)d_in[3];
  const float* qw     = (const float*)d_in[4];
  const float* kw     = (const float*)d_in[5];
  const float* vw     = (const float*)d_in[6];
  const float* pw     = (const float*)d_in[7];
  const float* pb     = (const float*)d_in[8];
  const float* srw    = (const float*)d_in[9];
  const float* srb    = (const float*)d_in[10];
  const float* gamma  = (const float*)d_in[11];
  const float* beta   = (const float*)d_in[12];
  const float* mean   = (const float*)d_in[13];
  const float* var    = (const float*)d_in[14];

  // workspace carve-up (~30.5 MiB)
  char* ws = (char*)d_ws;
  _Float16* xh  = (_Float16*)ws; ws += (size_t)B_SZ * N_TOK  * C_DIM * 2; // 8 MiB
  _Float16* xkh = (_Float16*)ws; ws += (size_t)B_SZ * NK_TOK * C_DIM * 2; // 2 MiB
  _Float16* qh  = (_Float16*)ws; ws += (size_t)B_SZ * N_TOK  * C_DIM * 2; // 8 MiB
  _Float16* kh  = (_Float16*)ws; ws += (size_t)B_SZ * NK_TOK * C_DIM * 2; // 2 MiB
  _Float16* vh  = (_Float16*)ws; ws += (size_t)B_SZ * NK_TOK * C_DIM * 2; // 2 MiB
  _Float16* oh  = (_Float16*)ws; ws += (size_t)B_SZ * N_TOK  * C_DIM * 2; // 8 MiB
  _Float16* wq  = (_Float16*)ws; ws += (size_t)C_DIM * C_DIM * 2;
  _Float16* wk  = (_Float16*)ws; ws += (size_t)C_DIM * C_DIM * 2;
  _Float16* wv  = (_Float16*)ws; ws += (size_t)C_DIM * C_DIM * 2;
  _Float16* wp  = (_Float16*)ws; ws += (size_t)C_DIM * C_DIM * 2;

  // f32 -> f16 converts
  cvt_f32_f16<<<4096, 256, 0, stream>>>(x,  xh, B_SZ * N_TOK * C_DIM);
  cvt_f32_f16<<<256,  256, 0, stream>>>(qw, wq, C_DIM * C_DIM);
  cvt_f32_f16<<<256,  256, 0, stream>>>(kw, wk, C_DIM * C_DIM);
  cvt_f32_f16<<<256,  256, 0, stream>>>(vw, wv, C_DIM * C_DIM);
  cvt_f32_f16<<<256,  256, 0, stream>>>(pw, wp, C_DIM * C_DIM);

  // spatial reduction (depthwise conv + BN) -> xk f16
  conv_bn<<<4096, 256, 0, stream>>>(x, srw, srb, gamma, beta, mean, var, xkh);

  // Q (pre-scaled by dh^-0.5), K, V GEMMs
  const float scale = 0.17677669529663687f;            // 32^-0.5
  dim3 gq(N_TOK * B_SZ / 16, C_DIM / 16 / 4);          // (1024, 4)
  dim3 gkv(NK_TOK * B_SZ / 16, C_DIM / 16 / 4);        // (256, 4)
  gemm_wmma<false><<<gq,  128, 0, stream>>>(xh,  wq, qh, nullptr, nullptr, scale);
  gemm_wmma<false><<<gkv, 128, 0, stream>>>(xkh, wk, kh, nullptr, nullptr, 1.0f);
  gemm_wmma<false><<<gkv, 128, 0, stream>>>(xkh, wv, vh, nullptr, nullptr, 1.0f);

  // fused attention (flash-style online softmax, rel_pos streamed once)
  dim3 ga(N_TOK / 64, B_SZ * HEADS);                   // (64, 32)
  attn_wmma<<<ga, 128, 0, stream>>>(qh, kh, vh, relpos, oh);

  // output projection with bias, f32 out
  gemm_wmma<true><<<gq, 128, 0, stream>>>(oh, wp, nullptr, (float*)d_out, pb, 1.0f);
}
